// SSBlock_27367531610164
// MI455X (gfx1250) — matrix operand order repaired and checked by reading, not compiled, not run on hardware
//
#include <hip/hip_runtime.h>
#include <hip/hip_bf16.h>

// ---------------------------------------------------------------------------
// Mamba-style selective-state-space block for MI455X (gfx1250, wave32, WMMA)
// D_MODEL=512, D_STATE=16, D_CONV=4, D_INNER=1024, B=16, SEQ=2048, M=32768
// ---------------------------------------------------------------------------

typedef __bf16 bf16;
typedef bf16  bfvec16 __attribute__((ext_vector_type(16)));
typedef float fvec8   __attribute__((ext_vector_type(8)));
typedef unsigned int uivec4 __attribute__((ext_vector_type(4)));

union Frag { bfvec16 v; uivec4 q[2]; };

static __device__ __forceinline__ fvec8 wmma_bf16(bfvec16 a, bfvec16 b, fvec8 c) {
  // v_wmma_f32_16x16x32_bf16: D = A(16x32) * B(32x16) + C(16x16 f32)
  return __builtin_amdgcn_wmma_f32_16x16x32_bf16(
      /*neg_a=*/false, a, /*neg_b=*/false, b,
      /*c_mod=*/(short)0, c, /*reuse_a=*/false, /*reuse_b=*/false);
}

// --------------------------- f32 -> bf16 convert ---------------------------
__global__ void cvt_bf16_kernel(const float* __restrict__ src,
                                bf16* __restrict__ dst, int n, int doClip) {
  int i = blockIdx.x * 256 + threadIdx.x;
  if (i < n) {
    float v = src[i];
    if (doClip) v = fminf(fmaxf(v, -10.f), 10.f);
    dst[i] = (bf16)v;
  }
}

// ------------------------------ WMMA GEMM ----------------------------------
// C[M,N] = A[M,K](bf16) @ W[N,K]^T(bf16)  (+ bias[N]) (+ clip(resid[M,N]))
// Block: 256 threads = 8 waves; block tile 128(M) x 64(N), BK = 32.
// Wave (wm = wid&3, wn = wid>>2) owns a 32x32 sub-tile = 2x2 WMMA tiles.
__global__ __launch_bounds__(256)
void wmma_gemm_kernel(const bf16* __restrict__ A, const bf16* __restrict__ W,
                      float* __restrict__ C, const float* __restrict__ bias,
                      const float* __restrict__ resid,
                      int M, int N, int K, int clipResid) {
  __shared__ __align__(16) unsigned char sA[128 * 80];  // 80B pitch: no bank camping
  __shared__ __align__(16) unsigned char sB[64 * 80];

  const int tid  = threadIdx.x;
  const int lane = tid & 31;
  const int half = lane >> 4;     // 0: lanes 0-15, 1: lanes 16-31
  const int lr   = lane & 15;
  const int wid  = tid >> 5;
  const int wm   = wid & 3;       // 0..3  (M groups of 32)
  const int wn   = wid >> 2;      // 0..1  (N groups of 32)

  const int mBase = blockIdx.y * 128;
  const int nBase = blockIdx.x * 64;

  // Per-lane LDS fragment byte offsets (ISA 7.12.2 bf16 layouts)
  int aOff[2], bOff[2];
#pragma unroll
  for (int i = 0; i < 2; ++i) aOff[i] = (wm * 32 + i * 16 + lr) * 80 + half * 16;
#pragma unroll
  for (int j = 0; j < 2; ++j) bOff[j] = (wn * 32 + j * 16 + lr) * 80 + half * 16;

  fvec8 acc[2][2] = {};

  for (int k0 = 0; k0 < K; k0 += 32) {
    // Stage A tile: 128 rows x 32 bf16 (64B/row), 512 b128 transfers
#pragma unroll
    for (int i = 0; i < 2; ++i) {
      int idx = i * 256 + tid;           // 0..511
      int row = idx >> 2, q = idx & 3;
      *(uivec4*)(sA + row * 80 + q * 16) =
          *(const uivec4*)(A + (size_t)(mBase + row) * K + k0 + q * 8);
    }
    // Stage B tile: 64 N-rows x 32 bf16, rows of W are already K-contiguous
    {
      int n = tid >> 2, q = tid & 3;
      uivec4 val = (uivec4)0;
      if (nBase + n < N)
        val = *(const uivec4*)(W + (size_t)(nBase + n) * K + k0 + q * 8);
      *(uivec4*)(sB + n * 80 + q * 16) = val;
    }
    __syncthreads();

    Frag aF[2], bF[2];
#pragma unroll
    for (int i = 0; i < 2; ++i) {
      aF[i].q[0] = *(const uivec4*)(sA + aOff[i]);        // K = 8*half .. +7
      aF[i].q[1] = *(const uivec4*)(sA + aOff[i] + 32);   // K = 16+8*half .. +7
    }
#pragma unroll
    for (int j = 0; j < 2; ++j) {
      bF[j].q[0] = *(const uivec4*)(sB + bOff[j]);        // K = 8*half .. +7
      bF[j].q[1] = *(const uivec4*)(sB + bOff[j] + 32);   // K = 16+8*half .. +7
    }
#pragma unroll
    for (int i = 0; i < 2; ++i)
#pragma unroll
      for (int j = 0; j < 2; ++j)
        acc[i][j] = wmma_bf16(aF[i].v, bF[j].v, acc[i][j]);
    __syncthreads();
  }

  // Epilogue: D-frag element e -> (m = tileM + e + 8*half, n = tileN + lr)
#pragma unroll
  for (int i = 0; i < 2; ++i) {
#pragma unroll
    for (int j = 0; j < 2; ++j) {
      int n = nBase + wn * 32 + j * 16 + lr;
      if (n < N) {
        int m0 = mBase + wm * 32 + i * 16 + half * 8;
        float bv = bias ? bias[n] : 0.f;
#pragma unroll
        for (int e = 0; e < 8; ++e) {
          int m = m0 + e;
          float v = acc[i][j][e] + bv;
          if (resid) {
            float r = resid[(size_t)m * N + n];
            if (clipResid) r = fminf(fmaxf(r, -10.f), 10.f);
            v += r;
          }
          C[(size_t)m * N + n] = v;
        }
      }
    }
  }
}

// ------------------- causal depthwise conv (width 4) + SiLU ----------------
// x_p = xr[:, 0:1024] of the [M,2048] GEMM1 output; emits bf16 xc.
__global__ void conv_silu_kernel(const float* __restrict__ xr,
                                 const float* __restrict__ cw,
                                 const float* __restrict__ cb,
                                 bf16* __restrict__ xc) {
  int idx = blockIdx.x * 256 + threadIdx.x;          // over M*1024
  int d = idx & 1023;
  int m = idx >> 10;
  int t = m & 2047;
  float acc = cb[d];
#pragma unroll
  for (int k = 0; k < 4; ++k) {
    int tt = t - 3 + k;
    if (tt >= 0)
      acc += cw[d * 4 + k] * xr[(size_t)(m - 3 + k) * 2048 + d];
  }
  float s = acc / (1.f + __expf(-acc));              // SiLU
  xc[idx] = (bf16)s;
}

// --------------------------- selective scan --------------------------------
// One thread per (b, d_inner) channel; 16 SSM states in registers.
// B_t/C_t (32 floats, uniform per (b,t)) broadcast through LDS.
// Fused epilogue: z = (y + u*D) * silu(res), emitted as bf16 for GEMM3.
__global__ __launch_bounds__(256)
void scan_kernel(const float* __restrict__ ssm, const float* __restrict__ dt,
                 const bf16* __restrict__ xc, const float* __restrict__ xr,
                 const float* __restrict__ Alog, const float* __restrict__ Dp,
                 bf16* __restrict__ z) {
  const int b = blockIdx.y;
  const int d = blockIdx.x * 256 + threadIdx.x;
  float A[16];
#pragma unroll
  for (int j = 0; j < 16; ++j) A[j] = Alog[j];
  const float Dv = Dp[d];
  float st[16] = {};
  __shared__ float sBC[32];

  for (int t = 0; t < 2048; ++t) {
    const int m = b * 2048 + t;
    if (threadIdx.x < 32) sBC[threadIdx.x] = ssm[(size_t)m * 32 + threadIdx.x];
    __syncthreads();
    float dtv = dt[(size_t)m * 1024 + d];
    float u   = (float)xc[(size_t)m * 1024 + d];
    float du  = dtv * u;
    float y = 0.f;
#pragma unroll
    for (int j = 0; j < 16; ++j) {
      float e = fminf(fmaxf(dtv * A[j], -5.f), 5.f);
      float dA = __expf(e);                           // v_exp_f32
      st[j] = dA * st[j] + du * sBC[j];
      y += st[j] * sBC[16 + j];
    }
    y += u * Dv;
    float r  = xr[(size_t)m * 2048 + 1024 + d];       // res half of GEMM1 out
    float sr = r / (1.f + __expf(-r));                // silu(res)
    z[(size_t)m * 1024 + d] = (bf16)(y * sr);
    __syncthreads();
  }
}

// ------------------------- in-place LayerNorm ------------------------------
__global__ __launch_bounds__(256)
void ln_kernel(float* __restrict__ out, const float* __restrict__ g,
               const float* __restrict__ bta) {
  const int row = blockIdx.x, tid = threadIdx.x;
  const size_t base = (size_t)row * 512;
  float v0 = out[base + tid], v1 = out[base + 256 + tid];
  __shared__ float red[256];
  red[tid] = v0 + v1;
  __syncthreads();
  for (int s = 128; s > 0; s >>= 1) {
    if (tid < s) red[tid] += red[tid + s];
    __syncthreads();
  }
  float mu = red[0] * (1.f / 512.f);
  __syncthreads();
  float d0 = v0 - mu, d1 = v1 - mu;
  red[tid] = d0 * d0 + d1 * d1;
  __syncthreads();
  for (int s = 128; s > 0; s >>= 1) {
    if (tid < s) red[tid] += red[tid + s];
    __syncthreads();
  }
  float rs = rsqrtf(red[0] * (1.f / 512.f) + 1e-5f);
  out[base + tid]       = d0 * rs * g[tid] + bta[tid];
  out[base + 256 + tid] = d1 * rs * g[tid + 256] + bta[tid + 256];
}

// ---------------------------------------------------------------------------
extern "C" void kernel_launch(void* const* d_in, const int* in_sizes, int n_in,
                              void* d_out, int out_size, void* d_ws, size_t ws_size,
                              hipStream_t stream) {
  (void)in_sizes; (void)n_in; (void)out_size; (void)ws_size;
  const float* x      = (const float*)d_in[0];   // [16,2048,512]
  const float* W_in   = (const float*)d_in[1];   // [2048,512]
  const float* conv_w = (const float*)d_in[2];   // [1024,1,4]
  const float* conv_b = (const float*)d_in[3];   // [1024]
  const float* W_x    = (const float*)d_in[4];   // [32,1024]
  const float* W_dt   = (const float*)d_in[5];   // [1024,1024]
  const float* b_dt   = (const float*)d_in[6];   // [1024]
  const float* A_log  = (const float*)d_in[7];   // [16]
  const float* Dp     = (const float*)d_in[8];   // [1024]
  const float* W_out  = (const float*)d_in[9];   // [512,1024]
  const float* ln_g   = (const float*)d_in[10];  // [512]
  const float* ln_b   = (const float*)d_in[11];  // [512]
  float* out = (float*)d_out;                    // [16,2048,512] f32

  constexpr int M = 16 * 2048;                   // 32768 rows

  char* ws = (char*)d_ws;
  size_t off = 0;
  auto alloc = [&](size_t bytes) -> char* {
    char* p = ws + off;
    off += (bytes + 255) & ~(size_t)255;
    return p;
  };
  bf16*  xbf    = (bf16*)alloc((size_t)M * 512 * 2);     //  32 MB
  bf16*  winbf  = (bf16*)alloc((size_t)2048 * 512 * 2);  //   2 MB
  bf16*  wdtbf  = (bf16*)alloc((size_t)1024 * 1024 * 2); //   2 MB
  bf16*  wxbf   = (bf16*)alloc((size_t)32 * 1024 * 2);   //  64 KB
  bf16*  woutbf = (bf16*)alloc((size_t)512 * 1024 * 2);  //   1 MB
  float* xr     = (float*)alloc((size_t)M * 2048 * 4);   // 256 MB (x_p | res)
  bf16*  xcbf   = (bf16*)alloc((size_t)M * 1024 * 2);    //  64 MB
  float* dtbuf  = (float*)alloc((size_t)M * 1024 * 4);   // 128 MB
  float* ssm    = (float*)alloc((size_t)M * 32 * 4);     //   4 MB
  bf16*  zbf    = (bf16*)alloc((size_t)M * 1024 * 2);    //  64 MB

  // 1) bf16 conversions (x clipped into xbf for GEMM1; GEMM3 re-clips the f32 x residual)
  cvt_bf16_kernel<<<(M * 512 + 255) / 256, 256, 0, stream>>>(x, xbf, M * 512, 1);
  cvt_bf16_kernel<<<(2048 * 512 + 255) / 256, 256, 0, stream>>>(W_in, winbf, 2048 * 512, 0);
  cvt_bf16_kernel<<<(1024 * 1024 + 255) / 256, 256, 0, stream>>>(W_dt, wdtbf, 1024 * 1024, 0);
  cvt_bf16_kernel<<<(32 * 1024 + 255) / 256, 256, 0, stream>>>(W_x, wxbf, 32 * 1024, 0);
  cvt_bf16_kernel<<<(512 * 1024 + 255) / 256, 256, 0, stream>>>(W_out, woutbf, 512 * 1024, 0);

  // 2) GEMM1: xr[M,2048] = clip(x) @ W_in^T
  wmma_gemm_kernel<<<dim3(2048 / 64, M / 128), 256, 0, stream>>>(
      xbf, winbf, xr, nullptr, nullptr, M, 2048, 512, 0);

  // 3) causal depthwise conv + SiLU -> xc (bf16)
  conv_silu_kernel<<<(M * 1024) / 256, 256, 0, stream>>>(xr, conv_w, conv_b, xcbf);

  // 4) GEMM2: dt[M,1024] = xc @ W_dt^T + b_dt ; GEMM2b: ssm[M,32] = xc @ W_x^T
  wmma_gemm_kernel<<<dim3(1024 / 64, M / 128), 256, 0, stream>>>(
      xcbf, wdtbf, dtbuf, b_dt, nullptr, M, 1024, 1024, 0);
  wmma_gemm_kernel<<<dim3(1, M / 128), 256, 0, stream>>>(
      xcbf, wxbf, ssm, nullptr, nullptr, M, 32, 1024, 0);

  // 5) selective scan (sequential over SEQ), fused * silu(res) -> z (bf16)
  scan_kernel<<<dim3(1024 / 256, 16), 256, 0, stream>>>(
      ssm, dtbuf, xcbf, xr, A_log, Dp, zbf);

  // 6) GEMM3: out[M,512] = z @ W_out^T + clip(x)
  wmma_gemm_kernel<<<dim3(512 / 64, M / 128), 256, 0, stream>>>(
      zbf, woutbf, out, nullptr, x, M, 512, 1024, 1);

  // 7) in-place LayerNorm over last dim (512)
  ln_kernel<<<M, 256, 0, stream>>>(out, ln_g, ln_b);
}
